// BuildGcn_3075196584644
// MI455X (gfx1250) — matrix operand order repaired and checked by reading, not compiled, not run on hardware
//
#include <hip/hip_runtime.h>
#include <hip/hip_bf16.h>

// ---------------------------------------------------------------------------
// GCN layer for MI455X (gfx1250, wave32):
//   deg/dinv -> normalized scatter-add (bandwidth/atomic bound, ~75us roofline)
//   -> 100000x128 @ 128x128 GEMM via v_wmma_f32_16x16x32_bf16 with split-bf16
//      (hi+lo) emulation for ~fp32 accuracy at matrix-core speed.
// ---------------------------------------------------------------------------

typedef __attribute__((ext_vector_type(16))) __bf16 v16bf;
typedef __attribute__((ext_vector_type(8)))  float  v8f;

#define CH 128

__global__ void __launch_bounds__(256) k_deg_init(float* __restrict__ deg, int N) {
    int i = blockIdx.x * 256 + threadIdx.x;
    if (i < N) deg[i] = 1.0f;  // self-loop contribution folded in
}

__global__ void __launch_bounds__(256) k_deg_count(const int* __restrict__ col,
                                                   float* __restrict__ deg, int E) {
    int e = blockIdx.x * 256 + threadIdx.x;
    if (e < E) atomicAdd(&deg[col[e]], 1.0f);
}

__global__ void __launch_bounds__(256) k_dinv(const float* __restrict__ deg,
                                              float* __restrict__ dinv, int N) {
    int i = blockIdx.x * 256 + threadIdx.x;
    if (i < N) {
        float d = deg[i];
        dinv[i] = (d > 0.0f) ? rsqrtf(d) : 0.0f;
    }
}

// agg[i][:] = dinv[i] * x[i][:]   (the self-loop message; no atomics needed)
__global__ void __launch_bounds__(256) k_self_init(const float* __restrict__ x,
                                                   const float* __restrict__ dinv,
                                                   float* __restrict__ agg, int N) {
    int idx = blockIdx.x * 256 + threadIdx.x;  // one thread per (node, 4 channels)
    int i  = idx >> 5;
    int c4 = idx & 31;
    if (i >= N) return;
    float s = dinv[i];
    float4 v = reinterpret_cast<const float4*>(x + (size_t)i * CH)[c4];
    float4 o; o.x = s * v.x; o.y = s * v.y; o.z = s * v.z; o.w = s * v.w;
    reinterpret_cast<float4*>(agg + (size_t)i * CH)[c4] = o;
}

// One wave32 per edge: 32 lanes x float4 = 128 channels.
// Reads x[src] coalesced (512B), 4 global_atomic_add_f32 per lane into agg[dst].
__global__ void __launch_bounds__(256) k_scatter(const float* __restrict__ x,
                                                 const int* __restrict__ row,
                                                 const int* __restrict__ col,
                                                 const float* __restrict__ dinv,
                                                 float* __restrict__ agg, int E) {
    int e = blockIdx.x * 8 + (threadIdx.x >> 5);
    if (e >= E) return;
    int lane = threadIdx.x & 31;
    int dst = row[e];
    int src = col[e];
    float s = dinv[src];
    float4 v = reinterpret_cast<const float4*>(x + (size_t)src * CH)[lane];
    float* ar = agg + (size_t)dst * CH + lane * 4;
    atomicAdd(ar + 0, s * v.x);
    atomicAdd(ar + 1, s * v.y);
    atomicAdd(ar + 2, s * v.z);
    atomicAdd(ar + 3, s * v.w);
}

__device__ __forceinline__ void splitf(float v, __bf16& hi, __bf16& lo) {
    __bf16 h = (__bf16)v;           // RNE truncation to bf16
    hi = h;
    lo = (__bf16)(v - (float)h);    // residual captures the next ~8 mantissa bits
}

// out = (dinv ⊙ agg) @ W^T + b
// Block = 256 threads = 8 waves; wave w computes the 16x16 tile at
// (m0 = 16*blockIdx.x, n0 = 16*w). K=128 -> 4 steps of 16x16x32 bf16 WMMA,
// 3 products per step (hi*hi + hi*lo + lo*hi) for ~fp32 accuracy.
__global__ void __launch_bounds__(256) k_gemm_wmma(const float* __restrict__ agg,
                                                   const float* __restrict__ dinv,
                                                   const float* __restrict__ W,   // [128][128] = W[n][k]
                                                   const float* __restrict__ bias,
                                                   float* __restrict__ out, int N) {
    int wave = threadIdx.x >> 5;
    int lane = threadIdx.x & 31;
    int l16  = lane & 15;
    int half = lane >> 4;

    int m0 = blockIdx.x * 16;
    int n0 = wave * 16;

    // A-side row for this lane (both halves load the same 16 rows, different K)
    int mA = m0 + l16;
    if (mA >= N) mA = N - 1;                 // clamp (stores are guarded below)
    float s = dinv[mA];
    const float* arow = agg + (size_t)mA * CH;
    const float* wrow = W + (size_t)(n0 + l16) * CH;  // B column n = n0+l16 reads W[n][*]

    v8f acc = {};

    #pragma unroll
    for (int kk = 0; kk < 4; ++kk) {
        int k0 = kk * 32;
        v16bf a_hi, a_lo, b_hi, b_lo;

        // A (16x32 bf16): element j in 0..15 -> K = half*16 + j (matches B)
        #pragma unroll
        for (int r = 0; r < 2; ++r) {
            int ka = k0 + half * 16 + r * 8;
            #pragma unroll
            for (int j = 0; j < 8; ++j) {
                __bf16 h, l;
                splitf(s * arow[ka + j], h, l);
                a_hi[r * 8 + j] = h;
                a_lo[r * 8 + j] = l;
            }
        }
        // B (32x16 bf16, contiguous K, same as A): lane holds column n; element j -> K = half*16 + j
        {
            int kb = k0 + half * 16;
            #pragma unroll
            for (int j = 0; j < 16; ++j) {
                __bf16 h, l;
                splitf(wrow[kb + j], h, l);
                b_hi[j] = h;
                b_lo[j] = l;
            }
        }

        acc = __builtin_amdgcn_wmma_f32_16x16x32_bf16(false, a_hi, false, b_hi,
                                                      (short)0, acc, false, false);
        acc = __builtin_amdgcn_wmma_f32_16x16x32_bf16(false, a_hi, false, b_lo,
                                                      (short)0, acc, false, false);
        acc = __builtin_amdgcn_wmma_f32_16x16x32_bf16(false, a_lo, false, b_hi,
                                                      (short)0, acc, false, false);
    }

    // C/D layout: acc[v] is (m = m0 + half*8 + v, n = n0 + l16)
    float bv = bias[n0 + l16];
    #pragma unroll
    for (int v = 0; v < 8; ++v) {
        int m = m0 + half * 8 + v;
        if (m < N) out[(size_t)m * CH + n0 + l16] = acc[v] + bv;
    }
}

extern "C" void kernel_launch(void* const* d_in, const int* in_sizes, int n_in,
                              void* d_out, int out_size, void* d_ws, size_t ws_size,
                              hipStream_t stream) {
    (void)n_in; (void)out_size; (void)ws_size;

    const float* x  = (const float*)d_in[0];        // [N,128]
    const int*   ei = (const int*)d_in[1];          // [2,E] flat: row then col
    const float* W  = (const float*)d_in[2];        // [128,128]
    const float* b  = (const float*)d_in[3];        // [128]
    float* out = (float*)d_out;                     // [N,128]

    const int N = in_sizes[0] / CH;
    const int E = in_sizes[1] / 2;
    const int* row = ei;        // destination
    const int* col = ei + E;    // source

    // Workspace layout: deg[N] | dinv[N] | agg[N*128]
    float* deg  = (float*)d_ws;
    float* dinv = deg + N;
    float* agg  = dinv + N;

    k_deg_init <<<(N + 255) / 256, 256, 0, stream>>>(deg, N);
    k_deg_count<<<(E + 255) / 256, 256, 0, stream>>>(col, deg, E);
    k_dinv     <<<(N + 255) / 256, 256, 0, stream>>>(deg, dinv, N);
    k_self_init<<<((N * 32) + 255) / 256, 256, 0, stream>>>(x, dinv, agg, N);
    k_scatter  <<<(E + 7) / 8, 256, 0, stream>>>(x, row, col, dinv, agg, E);
    k_gemm_wmma<<<(N + 15) / 16, 256, 0, stream>>>(agg, dinv, W, b, out, N);
}
